// InitialContraction_17978733101257
// MI455X (gfx1250) — matrix operand order repaired and checked by reading, not compiled, not run on hardware
//
#include <hip/hip_runtime.h>
#include <hip/hip_bf16.h>
#include <stdint.h>

// Problem dims
#define Bn   256
#define Cn   64
#define Yn   16
#define Zn   23
#define En   10
#define Kdim 368            // Y*Z (i,k contraction)
#define Kpad 384            // padded to multiple of 32 for 16x16x32 WMMA
#define Mdim 16384          // B*C
#define Ndim 4096           // Y^3

// GEMM tiling
#define BM  128
#define BN  128
#define KT  96              // K halves per LDS chunk (multiple of 32)
#define SA  104             // padded LDS row stride in halves (208B: conflict-free, 16B aligned)
#define NKC (Kpad / KT)     // 4 chunks

typedef __attribute__((ext_vector_type(16))) __bf16 v16bf;
typedef __attribute__((ext_vector_type(8)))  float  v8f;

union FragBF { uint4 q[2]; v16bf v; };
union AccF   { float f[8];  v8f  v; };

// Async global->LDS (gfx1250) if the toolchain exposes the builtins; else
// fall back to VGPR staging (global_load_b128 + ds_store_b128).
#if __has_builtin(__builtin_amdgcn_global_load_async_to_lds_b128)
#define HAVE_ASYNC_LDS 1
// Exact parameter types per the compiler diagnostic:
//   param0: int __attribute__((vector_size(16))) addrspace(1)*   (global src)
//   param1: ... addrspace(3)*                                    (LDS dst)
typedef int v4i_a __attribute__((vector_size(16)));
typedef __attribute__((address_space(1))) v4i_a* as1_v4i;
typedef __attribute__((address_space(3))) v4i_a* as3_v4i;

__device__ __forceinline__ void async_copy16(const uint4* g, uint4* l) {
  // Integer round-trip: AS1 keeps the 64-bit address; AS3 pointers are 32-bit
  // and a generic __shared__ pointer's low 32 bits are the LDS byte offset.
  __builtin_amdgcn_global_load_async_to_lds_b128(
      (as1_v4i)(uintptr_t)g,
      (as3_v4i)(uint32_t)(uintptr_t)l,
      0, 0);
}
#else
#define HAVE_ASYNC_LDS 0
#endif

__device__ __forceinline__ void wait_asynccnt0() {
#if __has_builtin(__builtin_amdgcn_s_wait_asynccnt)
  __builtin_amdgcn_s_wait_asynccnt(0);
#else
  asm volatile("s_wait_asynccnt 0x0" ::: "memory");
#endif
}

__device__ __forceinline__ unsigned short f32_to_bf16_rne(float f) {
  union { float f; unsigned int u; } cv; cv.f = f;
  unsigned int u = cv.u;
  u += 0x7FFFu + ((u >> 16) & 1u);          // round-to-nearest-even
  return (unsigned short)(u >> 16);
}

// --- Kernel 1: U (4096 x 368 f32, contiguous) -> Ubf (4096 x 384 bf16, zero-padded)
__global__ void pack_u_bf16(const float* __restrict__ U,
                            unsigned short* __restrict__ Ubf) {
  const int n = blockIdx.x;                 // 0..4095  (w*256 + x*16 + v)
  const int t = threadIdx.x;                // 0..383
  unsigned short v = 0;
  if (t < Kdim) v = f32_to_bf16_rne(U[n * Kdim + t]);
  Ubf[n * Kpad + t] = v;
}

// --- Kernel 2: A[m=(b,c)][kk=(i,k)] = (sum_e W[e,k,c]*attr[b,e]) * z[b,c,i], bf16, padded
__global__ void build_a_bf16(const float* __restrict__ W,      // (E,Z,C)
                             const float* __restrict__ z,      // (B,C,Y)
                             const float* __restrict__ attr,   // (B,E)
                             unsigned short* __restrict__ Abf) {
  const int m = blockIdx.x;                 // b*64 + c
  const int b = m >> 6;
  const int c = m & 63;
  const int t = threadIdx.x;                // 0..383
  unsigned short outv = 0;
  if (t < Kdim) {
    const int i = t / Zn;
    const int k = t - i * Zn;
    float wb = 0.f;
#pragma unroll
    for (int e = 0; e < En; ++e)
      wb += W[(e * Zn + k) * Cn + c] * attr[b * En + e];
    outv = f32_to_bf16_rne(wb * z[m * Yn + i]);
  }
  Abf[m * Kpad + t] = outv;
}

// --- Kernel 3: D(16384x4096,f32) = Abf(16384x384) * Ubf(4096x384)^T via bf16 WMMA
__global__ void __launch_bounds__(256)
wmma_gemm_bf16(const unsigned short* __restrict__ Abf,
               const unsigned short* __restrict__ Ubf,
               float* __restrict__ out) {
  __shared__ __align__(16) unsigned short As[BM * SA];
  __shared__ __align__(16) unsigned short Bs[BN * SA];

  const int tid  = threadIdx.x;
  const int lane = tid & 31;
  const int wave = tid >> 5;                 // 8 waves
  const int wm   = (wave & 3) * 32;          // 4 waves across M (32 rows each)
  const int wn   = (wave >> 2) * 64;         // 2 waves across N (64 cols each)

  const int m0 = blockIdx.y * BM;
  const int n0 = blockIdx.x * BN;

  const int lrow = lane & 15;
  const int lhi  = lane >> 4;                // 0 or 1 (upper half-wave)

  v8f acc[2][4];
  const v8f vzero = {0.f, 0.f, 0.f, 0.f, 0.f, 0.f, 0.f, 0.f};
#pragma unroll
  for (int mf = 0; mf < 2; ++mf)
#pragma unroll
    for (int nf = 0; nf < 4; ++nf)
      acc[mf][nf] = vzero;

  const uint4* Ag  = (const uint4*)Abf;      // Kpad/8 = 48 uint4 per row
  const uint4* Bg  = (const uint4*)Ubf;
  uint4* AsQ = (uint4*)As;                   // SA/8 = 13 uint4 per LDS row
  uint4* BsQ = (uint4*)Bs;

  for (int kc = 0; kc < NKC; ++kc) {
    __syncthreads();                         // protect prior chunk's reads
    // Stage 128 x 96 halves of A and B into LDS (6 x 16B per thread each)
#if HAVE_ASYNC_LDS
#pragma unroll
    for (int t = 0; t < 6; ++t) {
      const int idx = tid + t * 256;         // 0..1535
      const int r   = idx / 12;              // row 0..127
      const int j   = idx - r * 12;          // 16B chunk 0..11 within 96 halves
      async_copy16(&Ag[(m0 + r) * 48 + kc * 12 + j], &AsQ[r * 13 + j]);
      async_copy16(&Bg[(n0 + r) * 48 + kc * 12 + j], &BsQ[r * 13 + j]);
    }
#else
#pragma unroll
    for (int t = 0; t < 6; ++t) {
      const int idx = tid + t * 256;         // 0..1535
      const int r   = idx / 12;              // row 0..127
      const int j   = idx - r * 12;          // 16B chunk 0..11 within 96 halves
      AsQ[r * 13 + j] = Ag[(m0 + r) * 48 + kc * 12 + j];
      BsQ[r * 13 + j] = Bg[(n0 + r) * 48 + kc * 12 + j];
    }
#endif
    // Hint next chunk into cache (gfx1250 global_prefetch_b8)
    if (kc + 1 < NKC) {
      const int r = tid / 12, j = tid - r * 12;
      __builtin_prefetch(&Ag[(m0 + r) * 48 + (kc + 1) * 12 + j], 0, 0);
      __builtin_prefetch(&Bg[(n0 + r) * 48 + (kc + 1) * 12 + j], 0, 0);
    }
#if HAVE_ASYNC_LDS
    wait_asynccnt0();
#endif
    __syncthreads();

#pragma unroll
    for (int ks = 0; ks < KT; ks += 32) {
      FragBF afr[2], bfr[4];
      // A fragment (16x32 bf16): lanes 0-15 row M=lane, K halves {0..15};
      // lanes 16-31 same rows, K halves {16..31}  -> two 16B LDS reads
#pragma unroll
      for (int mf = 0; mf < 2; ++mf) {
        const unsigned short* p = &As[(wm + mf * 16 + lrow) * SA + ks + lhi * 16];
        afr[mf].q[0] = *(const uint4*)p;
        afr[mf].q[1] = *(const uint4*)(p + 8);
      }
      // B fragment (32x16 bf16): lane holds N=lane&15, 16 consecutive K starting
      // at (lane>>4)*16 -> one 32B LDS read (as two uint4)
#pragma unroll
      for (int nf = 0; nf < 4; ++nf) {
        const unsigned short* p = &Bs[(wn + nf * 16 + lrow) * SA + ks + lhi * 16];
        bfr[nf].q[0] = *(const uint4*)p;
        bfr[nf].q[1] = *(const uint4*)(p + 8);
      }
#pragma unroll
      for (int mf = 0; mf < 2; ++mf)
#pragma unroll
        for (int nf = 0; nf < 4; ++nf)
          acc[mf][nf] = __builtin_amdgcn_wmma_f32_16x16x32_bf16(
              false, afr[mf].v, false, bfr[nf].v,
              (short)0, acc[mf][nf], false, false);
    }
  }

  // C/D layout: VGPR r, lanes 0-15 -> M=r, N=lane; lanes 16-31 -> M=8+r, N=lane-16
  // Output is written once and never re-read: use non-temporal stores so the
  // streamed 268 MB doesn't evict the L2-resident Abf/Ubf operand tiles.
#pragma unroll
  for (int mf = 0; mf < 2; ++mf)
#pragma unroll
    for (int nf = 0; nf < 4; ++nf) {
      AccF a; a.v = acc[mf][nf];
      const int nG    = n0 + wn + nf * 16 + lrow;
      const int mBase = m0 + wm + mf * 16 + lhi * 8;
#pragma unroll
      for (int r = 0; r < 8; ++r)
        __builtin_nontemporal_store(a.f[r], &out[(size_t)(mBase + r) * Ndim + nG]);
    }
}

extern "C" void kernel_launch(void* const* d_in, const int* in_sizes, int n_in,
                              void* d_out, int out_size, void* d_ws, size_t ws_size,
                              hipStream_t stream) {
  const float* U    = (const float*)d_in[0];   // (16,16,16,16,23)
  const float* W    = (const float*)d_in[1];   // (10,23,64)
  const float* z    = (const float*)d_in[2];   // (256,64,16)
  const float* attr = (const float*)d_in[3];   // (256,10)
  float* out = (float*)d_out;                  // (256,64,16,16,16)

  unsigned short* Ubf = (unsigned short*)d_ws;                 // 4096*384*2  = 3.0 MB
  unsigned short* Abf = Ubf + (size_t)Ndim * Kpad;             // 16384*384*2 = 12.6 MB

  pack_u_bf16<<<Ndim, Kpad, 0, stream>>>(U, Ubf);
  build_a_bf16<<<Mdim, Kpad, 0, stream>>>(W, z, attr, Abf);

  dim3 grid(Ndim / BN, Mdim / BM);             // 32 x 128 workgroups
  wmma_gemm_bf16<<<grid, 256, 0, stream>>>(Abf, Ubf, out);
}
